// LargeKernel_47579647705207
// MI455X (gfx1250) — matrix operand order repaired and checked by reading, not compiled, not run on hardware
//
#include <hip/hip_runtime.h>

#define CDIM 384
#define NIMG 16
#define HH 56
#define WW 56
#define PLANE (HH*WW)          // 3136
#define PAD_L 15
#define PR 94                  // padded rows needed: 48+15+30 = 93
#define PITCH 120              // padded LDS row pitch (cols 0..111 used)
#define KL 31
#define KS 5

typedef __attribute__((ext_vector_type(16))) __bf16 v16bf;
typedef __attribute__((ext_vector_type(8)))  __bf16 v8bf;
typedef __attribute__((ext_vector_type(8)))  float  v8f;

__device__ __forceinline__ v8f wmma_bf16(v16bf a, v16bf b, v8f c) {
  // D = A(16x32 bf16) * B(32x16 bf16) + C(16x16 f32)
  return __builtin_amdgcn_wmma_f32_16x16x32_bf16(
      /*neg_a=*/false, a, /*neg_b=*/false, b,
      /*c_mod=*/(short)0, c, /*reuse_a=*/false, /*reuse_b=*/false);
}

// Build A operand for one K=32 chunk: per-lane comps 0-7 = P[row][base+0..7],
// comps 8-15 = P[row][base+16..23]  (K = (e>>3)*16 + half*8 + (e&7))
__device__ __forceinline__ v16bf mk_a(const __bf16* p) {
  union { v16bf v; v8bf h[2]; } u;
  u.h[0] = *(const v8bf*)(p);
  u.h[1] = *(const v8bf*)(p + 16);
  return u.v;
}

__launch_bounds__(256)
__global__ void dwconv_wmma_kernel(const float* __restrict__ x,
                                   const float* __restrict__ wL,
                                   const float* __restrict__ wS,
                                   float* __restrict__ yL,
                                   float* __restrict__ yS,
                                   float* __restrict__ partials) {
  __shared__ __align__(16) __bf16 sP[PR * PITCH];        // padded bf16 input plane
  __shared__ __align__(32) __bf16 sTL[KL * 2 * 32 * 16]; // Toeplitz large [dy][o][lane][comp]
  __shared__ __align__(32) __bf16 sTS[KS * 32 * 16];     // Toeplitz small [dy][lane][comp]
  __shared__ __bf16 sWL[KL * KL + 1];
  __shared__ __bf16 sWS[32];
  __shared__ float  sRed[8][4];

  const int tid = threadIdx.x;
  const int plane = blockIdx.x;                 // plane = n*CDIM + c
  const int c = plane % CDIM;
  const long planeBase = (long)plane * PLANE;

  // ---- phase 0: filter weights -> LDS (bf16) ----
  for (int i = tid; i < KL * KL; i += 256) sWL[i] = (__bf16)wL[c * KL * KL + i];
  if (tid < KS * KS) sWS[tid] = (__bf16)wS[c * KS * KS + tid];
  __syncthreads();

  // ---- phase 1: zero-padded input plane -> LDS (bf16) ----
  for (int i = tid; i < PR * PITCH; i += 256) {
    int r = i / PITCH, col = i - r * PITCH;
    float v = 0.f;
    int iy = r - PAD_L, ix = col - PAD_L;
    if ((unsigned)iy < (unsigned)HH && (unsigned)ix < (unsigned)WW)
      v = x[planeBase + iy * WW + ix];
    sP[i] = (__bf16)v;
  }
  // ---- phase 2a: large-filter Toeplitz in B-operand layout ----
  // B layout (16-bit, 32x16): lane L -> N = L%16, K = (comp>>3)*16 + (L>>4)*8 + (comp&7)
  for (int e = tid; e < KL * 1024; e += 256) {
    int comp = e & 15, lane = (e >> 4) & 31, o = (e >> 9) & 1, dy = e >> 10;
    int k = o * 32 + ((comp >> 3) << 4) + ((lane >> 4) << 3) + (comp & 7);
    int n = lane & 15;
    int d = k - n;                    // tap index dx
    sTL[e] = (d >= 0 && d < KL) ? sWL[dy * KL + d] : (__bf16)0.f;
  }
  // ---- phase 2b: small-filter Toeplitz (s = n + 5 + dx, keeps A 16B-aligned) ----
  for (int e = tid; e < KS * 512; e += 256) {
    int comp = e & 15, lane = (e >> 4) & 31, dy = e >> 9;
    int k = ((comp >> 3) << 4) + ((lane >> 4) << 3) + (comp & 7);
    int n = lane & 15;
    int d = k - n - 5;
    sTS[e] = (d >= 0 && d < KS) ? sWS[dy * KS + d] : (__bf16)0.f;
  }
  __syncthreads();

  // ---- phase 3: WMMA compute. 8 waves x 2 tiles cover the 4x4 tile grid ----
  const int wave = tid >> 5, lane = tid & 31;
  const int half = lane >> 4, m = lane & 15;
  const int y0 = (wave >> 1) * 16;
  const int x0a = (wave & 1) * 32;
  const int x0b = x0a + 16;

  v8f aL0 = {}, aL1 = {}, aS0 = {}, aS1 = {};

  for (int dy = 0; dy < KL; ++dy) {
    const __bf16* row = sP + (y0 + m + dy) * PITCH;
    v16bf b0 = *(const v16bf*)(sTL + ((dy * 2 + 0) * 32 + lane) * 16);
    v16bf b1 = *(const v16bf*)(sTL + ((dy * 2 + 1) * 32 + lane) * 16);
    v16bf a00 = mk_a(row + x0a + half * 8);
    v16bf a01 = mk_a(row + x0a + 32 + half * 8);
    aL0 = wmma_bf16(a00, b0, aL0);
    aL0 = wmma_bf16(a01, b1, aL0);
    v16bf a10 = mk_a(row + x0b + half * 8);
    v16bf a11 = mk_a(row + x0b + 32 + half * 8);
    aL1 = wmma_bf16(a10, b0, aL1);
    aL1 = wmma_bf16(a11, b1, aL1);
  }
  for (int dy = 0; dy < KS; ++dy) {
    const __bf16* row = sP + (y0 + m + 13 + dy) * PITCH;
    v16bf bs = *(const v16bf*)(sTS + (dy * 32 + lane) * 16);
    v16bf a0 = mk_a(row + x0a + 8 + half * 8);
    v16bf a1 = mk_a(row + x0b + 8 + half * 8);
    aS0 = wmma_bf16(a0, bs, aS0);
    aS1 = wmma_bf16(a1, bs, aS1);
  }

  // ---- phase 4: store y_L / y_S + deterministic plane reduction (sum, sumsq) ----
  float s0 = 0.f, s1 = 0.f, s2 = 0.f, s3 = 0.f;
  for (int t = 0; t < 2; ++t) {
    int xt = t ? x0b : x0a;
    v8f L = t ? aL1 : aL0;
    v8f S = t ? aS1 : aS0;
    int xx = xt + m;                        // N = lane%16
    for (int r = 0; r < 8; ++r) {
      int yy = y0 + r + half * 8;           // M = r + 8*(lane/16)
      if (yy < HH && xx < WW) {
        long idx = planeBase + yy * WW + xx;
        float lv = L[r], sv = S[r];
        yL[idx] = lv; yS[idx] = sv;
        s0 += lv; s1 += lv * lv; s2 += sv; s3 += sv * sv;
      }
    }
  }
  for (int off = 16; off > 0; off >>= 1) {
    s0 += __shfl_xor(s0, off, 32);
    s1 += __shfl_xor(s1, off, 32);
    s2 += __shfl_xor(s2, off, 32);
    s3 += __shfl_xor(s3, off, 32);
  }
  if (lane == 0) { sRed[wave][0] = s0; sRed[wave][1] = s1; sRed[wave][2] = s2; sRed[wave][3] = s3; }
  __syncthreads();
  if (tid == 0) {
    float t0 = 0.f, t1 = 0.f, t2 = 0.f, t3 = 0.f;
    for (int w2 = 0; w2 < 8; ++w2) { t0 += sRed[w2][0]; t1 += sRed[w2][1]; t2 += sRed[w2][2]; t3 += sRed[w2][3]; }
    partials[plane * 4 + 0] = t0; partials[plane * 4 + 1] = t1;
    partials[plane * 4 + 2] = t2; partials[plane * 4 + 3] = t3;
  }
}

__global__ void bn_coeff_kernel(const float* __restrict__ partials,
                                const float* __restrict__ gL, const float* __restrict__ bL,
                                const float* __restrict__ gS, const float* __restrict__ bS,
                                float* __restrict__ coef) {
  int c = blockIdx.x * blockDim.x + threadIdx.x;
  if (c >= CDIM) return;
  float s0 = 0.f, s1 = 0.f, s2 = 0.f, s3 = 0.f;
  for (int n = 0; n < NIMG; ++n) {           // fixed-order => deterministic
    const float* p = partials + (long)(n * CDIM + c) * 4;
    s0 += p[0]; s1 += p[1]; s2 += p[2]; s3 += p[3];
  }
  const float invN = 1.f / (float)(NIMG * PLANE);
  float mL = s0 * invN, vL = s1 * invN - mL * mL;
  float mS = s2 * invN, vS = s3 * invN - mS * mS;
  float iL = rsqrtf(vL + 1e-5f), iS = rsqrtf(vS + 1e-5f);
  float aL = gL[c] * iL, aS = gS[c] * iS;
  coef[c * 4 + 0] = aL;
  coef[c * 4 + 1] = bL[c] - mL * aL;
  coef[c * 4 + 2] = aS;
  coef[c * 4 + 3] = bS[c] - mS * aS;
}

__launch_bounds__(256)
__global__ void combine_kernel(float* __restrict__ out, const float* __restrict__ yS,
                               const float* __restrict__ coef) {
  long i = ((long)blockIdx.x * blockDim.x + threadIdx.x) * 4;   // total divisible by 4
  int c = (int)((i / PLANE) % CDIM);                            // PLANE%4==0 -> same c for all 4
  const float aL = coef[c * 4 + 0], cL = coef[c * 4 + 1];
  const float aS = coef[c * 4 + 2], cS = coef[c * 4 + 3];
  float4 L = *(const float4*)(out + i);
  float4 S = *(const float4*)(yS + i);
  float4 r;
  r.x = aL * L.x + cL + aS * S.x + cS;
  r.y = aL * L.y + cL + aS * S.y + cS;
  r.z = aL * L.z + cL + aS * S.z + cS;
  r.w = aL * L.w + cL + aS * S.w + cS;
  *(float4*)(out + i) = r;
}

extern "C" void kernel_launch(void* const* d_in, const int* in_sizes, int n_in,
                              void* d_out, int out_size, void* d_ws, size_t ws_size,
                              hipStream_t stream) {
  const float* x  = (const float*)d_in[0];
  const float* wL = (const float*)d_in[1];
  const float* gL = (const float*)d_in[2];
  const float* bL = (const float*)d_in[3];
  const float* wS = (const float*)d_in[4];
  const float* gS = (const float*)d_in[5];
  const float* bS = (const float*)d_in[6];
  float* out = (float*)d_out;

  const long total = (long)NIMG * CDIM * PLANE;   // 19,267,584
  float* ws       = (float*)d_ws;
  float* yS       = ws;                           // total floats
  float* partials = ws + total;                   // 6144*4 floats
  float* coef     = partials + (long)NIMG * CDIM * 4;  // 384*4 floats

  dwconv_wmma_kernel<<<NIMG * CDIM, 256, 0, stream>>>(x, wL, wS, out, yS, partials);
  bn_coeff_kernel<<<(CDIM + 127) / 128, 128, 0, stream>>>(partials, gL, bL, gS, bS, coef);
  combine_kernel<<<(int)(total / 4 / 256), 256, 0, stream>>>(out, yS, coef);
}
